// SelfAttention_40647570489761
// MI455X (gfx1250) — matrix operand order repaired and checked by reading, not compiled, not run on hardware
//
#include <hip/hip_runtime.h>

// Problem constants (match reference)
#define BB 4
#define CC 1024
#define LL 2048
#define HH 16
#define DD 64

typedef __attribute__((ext_vector_type(16))) __bf16          v16bf;
typedef __attribute__((ext_vector_type(8)))  float           v8f;
typedef __attribute__((ext_vector_type(8)))  unsigned short  u16x8;
typedef __attribute__((ext_vector_type(16))) unsigned short  u16x16;

// fp32 -> bf16, round-to-nearest-even
__device__ __forceinline__ unsigned short f2bf(float f) {
  unsigned int u = __float_as_uint(f);
  u += 0x7FFFu + ((u >> 16) & 1u);
  return (unsigned short)(u >> 16);
}

__device__ __forceinline__ v16bf mkfrag(u16x8 lo, u16x8 hi) {
  u16x16 u = __builtin_shufflevector(lo, hi, 0,1,2,3,4,5,6,7,8,9,10,11,12,13,14,15);
  return __builtin_bit_cast(v16bf, u);
}

__device__ __forceinline__ v8f wmma_bf16(v16bf a, v16bf b, v8f c) {
  return __builtin_amdgcn_wmma_f32_16x16x32_bf16(false, a, false, b, (short)0, c, false, false);
}

// ---------------------------------------------------------------------------
// Kernel 1: Wm (2C x C) ++ Wq (C x C) -> bf16 Wall (3C x C), row-major
// ---------------------------------------------------------------------------
__global__ void convert_w(const float* __restrict__ Wm, const float* __restrict__ Wq,
                          unsigned short* __restrict__ Wall) {
  int idx = blockIdx.x * blockDim.x + threadIdx.x;
  if (idx >= 3 * CC * CC) return;
  float v = (idx < 2 * CC * CC) ? Wm[idx] : Wq[idx - 2 * CC * CC];
  Wall[idx] = f2bf(v);
}

// ---------------------------------------------------------------------------
// Kernel 2: projections.  Out(o,l) = sum_c Wall[o][c] * X[b][c][l]
// Channel routing: o<C -> K[B,H,L,D];  C<=o<2C -> V[B,H,D,L];
//                  o>=2C -> Q[B,H,L,D] scaled by D^-1/2.
// Block = 256 thr = 8 waves; wave w owns rows [o0+16w, o0+16w+16), cols [l0,l0+16).
// ---------------------------------------------------------------------------
__global__ void __launch_bounds__(256)
proj_kernel(const float* __restrict__ X, const unsigned short* __restrict__ Wall,
            unsigned short* __restrict__ Qbf, unsigned short* __restrict__ Kbf,
            unsigned short* __restrict__ Vbf) {
  __shared__ unsigned short Xt[16 * 40];   // X tile transposed [l][c], padded stride

  const int b     = blockIdx.z;
  const int l0    = blockIdx.x * 16;
  const int o0    = blockIdx.y * 128;
  const int t     = threadIdx.x;
  const int w     = t >> 5;
  const int lane  = t & 31;
  const int laneR = lane & 15;
  const int hi    = lane >> 4;      // wave half
  const int h8    = hi * 8;

  const int orow = o0 + w * 16 + laneR;     // A-matrix row for this lane

  v8f acc = {0.f,0.f,0.f,0.f,0.f,0.f,0.f,0.f};

  for (int k0 = 0; k0 < CC; k0 += 32) {
    // stage X[b][k0..k0+31][l0..l0+15] -> Xt[l][k] as bf16 (transpose in LDS)
    #pragma unroll
    for (int i = 0; i < 2; ++i) {
      int e  = t + i * 256;
      int kk = e >> 4;
      int ll = e & 15;
      float v = X[((size_t)b * CC + (size_t)(k0 + kk)) * LL + l0 + ll];
      Xt[ll * 40 + kk] = f2bf(v);
    }
    __syncthreads();

    // A fragment: lane<16 holds K {0..7,16..23}, lane>=16 holds {8..15,24..31}
    const unsigned short* ap = Wall + (size_t)orow * CC + k0 + h8;
    v16bf A = mkfrag(*(const u16x8*)ap, *(const u16x8*)(ap + 16));

    // B fragment: lane n = column l0+laneR; lane<16 holds K {0..7,16..23}, lane>=16 holds {8..15,24..31}
    const unsigned short* bp = Xt + laneR * 40 + h8;
    v16bf Bf = mkfrag(*(const u16x8*)bp, *(const u16x8*)(bp + 16));

    acc = wmma_bf16(A, Bf, acc);
    __syncthreads();
  }

  // D layout: lane holds column n=laneR, rows hi*8 + (0..7)  -> contiguous o
  const int l     = l0 + laneR;
  const int obase = o0 + w * 16 + h8;
  if (obase < CC) {                                 // K region
    int h = obase >> 6, d = obase & 63;
    unsigned short* kp = Kbf + (((size_t)(b * HH + h) * LL) + l) * DD + d;
    u16x8 s;
    #pragma unroll
    for (int r = 0; r < 8; ++r) s[r] = f2bf(acc[r]);
    *(u16x8*)kp = s;
  } else if (obase < 2 * CC) {                      // V region, stored [h][d][l]
    int od = obase - CC;
    int h = od >> 6, d0 = od & 63;
    #pragma unroll
    for (int r = 0; r < 8; ++r)
      Vbf[(((size_t)(b * HH + h) * DD) + d0 + r) * LL + l] = f2bf(acc[r]);
  } else {                                          // Q region, pre-scaled
    int oq = obase - 2 * CC;
    int h = oq >> 6, d = oq & 63;
    unsigned short* qp = Qbf + (((size_t)(b * HH + h) * LL) + l) * DD + d;
    u16x8 s;
    #pragma unroll
    for (int r = 0; r < 8; ++r) s[r] = f2bf(acc[r] * 0.125f);   // 64^-0.5
    *(u16x8*)qp = s;
  }
}

// ---------------------------------------------------------------------------
// Kernel 3: flash attention.  One wave per 16-row q-tile, 32-key blocks.
// ---------------------------------------------------------------------------
__global__ void __launch_bounds__(256)
attn_kernel(const unsigned short* __restrict__ Qbf, const unsigned short* __restrict__ Kbf,
            const unsigned short* __restrict__ Vbf, const int* __restrict__ mask,
            float* __restrict__ out) {
  __shared__ unsigned short Pld[8 * 16 * 40];   // per-wave P transpose scratch

  const int b     = blockIdx.z;
  const int hh    = blockIdx.y;
  const int t     = threadIdx.x;
  const int w     = t >> 5;
  const int lane  = t & 31;
  const int laneR = lane & 15;
  const int hi    = lane >> 4;
  const int h8    = hi * 8;
  const int q0    = (blockIdx.x * 8 + w) * 16;

  const unsigned short* Qh = Qbf + (size_t)(b * HH + hh) * LL * DD;
  const unsigned short* Kh = Kbf + (size_t)(b * HH + hh) * LL * DD;
  const unsigned short* Vh = Vbf + (size_t)(b * HH + hh) * DD * LL;
  const int* mrow = mask + b * LL;
  unsigned short* P = Pld + w * (16 * 40);

  // Q fragments for d-chunks 0..31 and 32..63 (reused across all key blocks)
  v16bf qa[2];
  {
    const unsigned short* qp = Qh + (size_t)(q0 + laneR) * DD;
    #pragma unroll
    for (int dc = 0; dc < 2; ++dc) {
      const unsigned short* p = qp + dc * 32 + h8;
      qa[dc] = mkfrag(*(const u16x8*)p, *(const u16x8*)(p + 16));
    }
  }

  v8f O[4];
  #pragma unroll
  for (int i = 0; i < 4; ++i) O[i] = (v8f){0.f,0.f,0.f,0.f,0.f,0.f,0.f,0.f};
  float rm[8], rs[8];
  #pragma unroll
  for (int r = 0; r < 8; ++r) { rm[r] = -3.0e38f; rs[r] = 0.f; }

  for (int kb = 0; kb < LL; kb += 32) {
    if (kb + 32 < LL) {   // prefetch next key/value tiles
      __builtin_prefetch(Kh + (size_t)(kb + 32 + laneR) * DD, 0, 3);
      __builtin_prefetch(Vh + (size_t)laneR * LL + kb + 32, 0, 3);
    }

    v8f S0 = {0.f,0.f,0.f,0.f,0.f,0.f,0.f,0.f};
    v8f S1 = {0.f,0.f,0.f,0.f,0.f,0.f,0.f,0.f};
    // S = Q * K^T  (B-frag col = key, rows = d, contiguous in K[B,H,L,D])
    #pragma unroll
    for (int dc = 0; dc < 2; ++dc) {
      const unsigned short* kp0 = Kh + (size_t)(kb + laneR) * DD + dc * 32 + h8;
      S0 = wmma_bf16(qa[dc], mkfrag(*(const u16x8*)kp0, *(const u16x8*)(kp0 + 16)), S0);
      const unsigned short* kp1 = Kh + (size_t)(kb + 16 + laneR) * DD + dc * 32 + h8;
      S1 = wmma_bf16(qa[dc], mkfrag(*(const u16x8*)kp1, *(const u16x8*)(kp1 + 16)), S1);
    }

    // key mask (per lane = per column); branchless select
    const int m0 = mrow[kb + laneR];
    const int m1 = mrow[kb + 16 + laneR];
    #pragma unroll
    for (int r = 0; r < 8; ++r) {
      S0[r] = m0 ? S0[r] : -1e30f;
      S1[r] = m1 ? S1[r] : -1e30f;
    }

    // online softmax; rows 0..7 in lanes 0..15, rows 8..15 in lanes 16..31
    #pragma unroll
    for (int r = 0; r < 8; ++r) {
      float s0 = S0[r], s1 = S1[r];
      float mx = fmaxf(s0, s1);
      #pragma unroll
      for (int off = 1; off < 16; off <<= 1)
        mx = fmaxf(mx, __shfl_xor(mx, off, 32));
      float mnew  = fmaxf(rm[r], mx);
      float alpha = __expf(rm[r] - mnew);
      float p0 = __expf(s0 - mnew);
      float p1 = __expf(s1 - mnew);
      float sum = p0 + p1;
      #pragma unroll
      for (int off = 1; off < 16; off <<= 1)
        sum += __shfl_xor(sum, off, 32);
      rs[r] = rs[r] * alpha + sum;
      rm[r] = mnew;
      #pragma unroll
      for (int i = 0; i < 4; ++i) O[i][r] *= alpha;
      S0[r] = p0; S1[r] = p1;
    }

    // P (16x32, D-layout) -> LDS -> A-fragment layout; per-wave scratch, and
    // same-wave LDS ops are processed in order by the DS unit.
    #pragma unroll
    for (int r = 0; r < 8; ++r) {
      P[(h8 + r) * 40 + laneR]      = f2bf(S0[r]);
      P[(h8 + r) * 40 + 16 + laneR] = f2bf(S1[r]);
    }
    __builtin_amdgcn_fence(__ATOMIC_ACQ_REL, "wavefront");
    const unsigned short* pp = P + laneR * 40 + h8;
    v16bf pa = mkfrag(*(const u16x8*)pp, *(const u16x8*)(pp + 16));

    // O += P * V  (B-frag col = d, rows = key, contiguous in V[B,H,D,L])
    #pragma unroll
    for (int dt = 0; dt < 4; ++dt) {
      const unsigned short* vp = Vh + (size_t)(dt * 16 + laneR) * LL + kb + h8;
      O[dt] = wmma_bf16(pa, mkfrag(*(const u16x8*)vp, *(const u16x8*)(vp + 16)), O[dt]);
    }
  }

  // normalize and store to out[B,C,L]; lane holds 8 consecutive l per d-column
  float rinv[8];
  #pragma unroll
  for (int r = 0; r < 8; ++r) rinv[r] = 1.0f / rs[r];
  #pragma unroll
  for (int dt = 0; dt < 4; ++dt) {
    int c = hh * DD + dt * 16 + laneR;
    float* op = out + ((size_t)b * CC + c) * LL + q0 + h8;
    float4 lo4 = make_float4(O[dt][0]*rinv[0], O[dt][1]*rinv[1],
                             O[dt][2]*rinv[2], O[dt][3]*rinv[3]);
    float4 hi4 = make_float4(O[dt][4]*rinv[4], O[dt][5]*rinv[5],
                             O[dt][6]*rinv[6], O[dt][7]*rinv[7]);
    *(float4*)op       = lo4;
    *(float4*)(op + 4) = hi4;
  }
}

// ---------------------------------------------------------------------------
extern "C" void kernel_launch(void* const* d_in, const int* in_sizes, int n_in,
                              void* d_out, int out_size, void* d_ws, size_t ws_size,
                              hipStream_t stream) {
  const float* queries = (const float*)d_in[0];   // [B,C,L] fp32
  const int*   mask    = (const int*)d_in[1];     // [B,L]
  const float* Wm      = (const float*)d_in[2];   // [2C,C]
  const float* Wq      = (const float*)d_in[3];   // [C,C]
  float*       out     = (float*)d_out;           // [B,C,L]

  // workspace layout (~54 MB total)
  char* ws = (char*)d_ws;
  unsigned short* Wall = (unsigned short*)ws;  ws += (size_t)3 * CC * CC * 2;
  unsigned short* Qbf  = (unsigned short*)ws;  ws += (size_t)BB * CC * LL * 2;
  unsigned short* Kbf  = (unsigned short*)ws;  ws += (size_t)BB * CC * LL * 2;
  unsigned short* Vbf  = (unsigned short*)ws;

  convert_w<<<(3 * CC * CC) / 256, 256, 0, stream>>>(Wm, Wq, Wall);

  dim3 pg(LL / 16, (3 * CC) / 128, BB);
  proj_kernel<<<pg, 256, 0, stream>>>(queries, Wall, Qbf, Kbf, Vbf);

  dim3 ag(LL / (16 * 8), HH, BB);
  attn_kernel<<<ag, 256, 0, stream>>>(Qbf, Kbf, Vbf, mask, out);
}
